// RetinaNet_47545287967116
// MI455X (gfx1250) — matrix operand order repaired and checked by reading, not compiled, not run on hardware
//
#include <hip/hip_runtime.h>
#include <hip/hip_fp16.h>

typedef __attribute__((ext_vector_type(16))) _Float16 v16h;
typedef __attribute__((ext_vector_type(8)))  float    v8f;
typedef __attribute__((__vector_size__(16))) int      i32x4;

union Frag {
  v16h h;
  unsigned u[8];
  uint4 q[2];
};

#define NEGINF (-1.0e30f)
#define MAX_CINP 320

// ---------------- layout conversion / packing ----------------

__global__ void k_nchw_to_nhwc_h(const float* __restrict__ x, _Float16* __restrict__ y,
                                 int C, int Cp, int HW) {
  int idx = blockIdx.x * blockDim.x + threadIdx.x;
  int total = HW * Cp;
  if (idx >= total) return;
  int p = idx / Cp, c = idx - p * Cp;
  float v = (c < C) ? x[(long)c * HW + p] : 0.f;
  y[idx] = (_Float16)v;
}

// w: [Cout,Cin,K,K] f32  ->  [K*K][Coutp][Cinp] f16 (zero padded)
__global__ void k_pack_w(const float* __restrict__ w, _Float16* __restrict__ out,
                         int Cout, int Coutp, int Cin, int Cinp, int Ksz) {
  int idx = blockIdx.x * blockDim.x + threadIdx.x;
  int KK = Ksz * Ksz;
  int plane = Coutp * Cinp;
  int total = KK * plane;
  if (idx >= total) return;
  int kpos = idx / plane;
  int rem = idx - kpos * plane;
  int co = rem / Cinp;
  int ci = rem - co * Cinp;
  float v = 0.f;
  if (co < Cout && ci < Cin) {
    int kh = kpos / Ksz, kw = kpos - kh * Ksz;
    v = w[(((long)co * Cin + ci) * Ksz + kh) * Ksz + kw];
  }
  out[idx] = (_Float16)v;
}

// ---------------- implicit-GEMM conv via WMMA f16 ----------------
// x: NHWC f16 [IH,IW,Cinp]; wp: [K*K][Coutp][Cinp] f16; y: NHWC f16 [OH,OW,Coutp]
// 128 threads = 4 waves. Block tile = 64 pixels x 64 out-channels.
// Weights for the current kernel position are staged in LDS (async DMA when
// available) and shared by all 4 waves; each wave owns 16 pixels.
__global__ __launch_bounds__(128)
void k_conv_wmma(const _Float16* __restrict__ x, const _Float16* __restrict__ wp,
                 const float* __restrict__ bias, _Float16* __restrict__ y,
                 int IH, int IW, int Cinp, int OH, int OW, int Cout, int Coutp,
                 int Ksz, int stride, int pad, int relu) {
  __shared__ __align__(16) _Float16 wtile[64 * MAX_CINP];  // 40 KB

  const int tid = threadIdx.x;
  const int wv = tid >> 5;
  const int lane = tid & 31;
  const int hh = lane >> 4;        // lane half: K sub-range selector in A/B layouts
  const int r = lane & 15;         // A row / B column within 16-tile
  const int P = OH * OW;
  const int pBase = blockIdx.x * 64 + wv * 16;
  const int coBase = blockIdx.y << 6;

  int p = pBase + r;
  if (p >= P) p = P - 1;           // P is always a multiple of 64 here; safety clamp
  const int oh = p / OW;
  const int ow = p - oh * OW;

  v8f z = {0.f, 0.f, 0.f, 0.f, 0.f, 0.f, 0.f, 0.f};
  v8f acc0 = z, acc1 = z, acc2 = z, acc3 = z;

  const int KK = Ksz * Ksz;
  const int tile16 = (64 * Cinp) >> 3;  // # of 16-byte chunks in the LDS weight tile
  for (int kpos = 0; kpos < KK; ++kpos) {
    // ---- stage 64 x Cinp weight tile for this kernel position into LDS ----
    const _Float16* gw = wp + ((long)kpos * Coutp + coBase) * Cinp;
#if defined(__gfx1250__) && __has_builtin(__builtin_amdgcn_global_load_async_to_lds_b128) && __has_builtin(__builtin_amdgcn_s_wait_asynccnt)
    {
      __attribute__((address_space(1))) char* gsrc =
          (__attribute__((address_space(1))) char*)(unsigned long long)(const char*)gw;
      __attribute__((address_space(3))) char* ldst =
          (__attribute__((address_space(3))) char*)wtile;
      for (int t = tid; t < tile16; t += 128) {
        __builtin_amdgcn_global_load_async_to_lds_b128(
            (__attribute__((address_space(1))) i32x4*)(gsrc + (long)t * 16),
            (__attribute__((address_space(3))) i32x4*)(ldst + (long)t * 16), 0, 0);
      }
      __builtin_amdgcn_s_wait_asynccnt(0);
    }
#else
    {
      const uint4* gs = (const uint4*)gw;
      uint4* ls = (uint4*)wtile;
      for (int t = tid; t < tile16; t += 128) ls[t] = gs[t];
    }
#endif
    __syncthreads();

    const int kh = kpos / Ksz;
    const int kw = kpos - kh * Ksz;
    const int ih = oh * stride - pad + kh;
    const int iw = ow * stride - pad + kw;
    const bool inb = (ih >= 0) && (ih < IH) && (iw >= 0) && (iw < IW);
    const _Float16* xrow = x + (long)(ih * IW + iw) * Cinp;

    for (int cc = 0; cc < Cinp; cc += 32) {
      // A fragment (16x32 f16): per lane one contiguous 32B run
      // halves [cc + hh*16, +16)
      Frag a;
      uint4 z4 = {0u, 0u, 0u, 0u};
      a.q[0] = inb ? *(const uint4*)(xrow + cc + hh * 16) : z4;
      a.q[1] = inb ? *(const uint4*)(xrow + cc + hh * 16 + 8) : z4;

      // B fragments (32x16 f16) from LDS: per lane one contiguous 32B run
      // at halves [co*Cinp + cc + hh*16, +16)  -> two ds_load_b128
      Frag b0, b1, b2, b3;
      {
        const uint4* bp0 = (const uint4*)(wtile + ((r + 0)  * Cinp + cc + hh * 16));
        const uint4* bp1 = (const uint4*)(wtile + ((r + 16) * Cinp + cc + hh * 16));
        const uint4* bp2 = (const uint4*)(wtile + ((r + 32) * Cinp + cc + hh * 16));
        const uint4* bp3 = (const uint4*)(wtile + ((r + 48) * Cinp + cc + hh * 16));
        b0.q[0] = bp0[0]; b0.q[1] = bp0[1];
        b1.q[0] = bp1[0]; b1.q[1] = bp1[1];
        b2.q[0] = bp2[0]; b2.q[1] = bp2[1];
        b3.q[0] = bp3[0]; b3.q[1] = bp3[1];
      }
      acc0 = __builtin_amdgcn_wmma_f32_16x16x32_f16(false, a.h, false, b0.h, (short)0, acc0, false, false);
      acc1 = __builtin_amdgcn_wmma_f32_16x16x32_f16(false, a.h, false, b1.h, (short)0, acc1, false, false);
      acc2 = __builtin_amdgcn_wmma_f32_16x16x32_f16(false, a.h, false, b2.h, (short)0, acc2, false, false);
      acc3 = __builtin_amdgcn_wmma_f32_16x16x32_f16(false, a.h, false, b3.h, (short)0, acc3, false, false);
    }
    __syncthreads();  // protect wtile before next kpos overwrites it
  }

  // Epilogue: D layout — VGPR j holds row (j + 8*laneHalf), col = lane&15
#pragma unroll
  for (int t = 0; t < 4; ++t) {
    v8f acc = (t == 0) ? acc0 : (t == 1) ? acc1 : (t == 2) ? acc2 : acc3;
    int co = coBase + t * 16 + r;
    float bv = (co < Cout) ? bias[co] : 0.f;
#pragma unroll
    for (int j = 0; j < 8; ++j) {
      float val = acc[j] + bv;
      if (relu) val = fmaxf(val, 0.f);
      int pp = pBase + j + hh * 8;
      if (pp < P) y[(long)pp * Coutp + co] = (_Float16)val;
    }
  }
}

// ---------------- elementwise ----------------

__global__ void k_up2_add(const _Float16* __restrict__ a, const _Float16* __restrict__ b,
                          _Float16* __restrict__ d, int OH, int OW, int C) {
  int idx = blockIdx.x * blockDim.x + threadIdx.x;
  int total = OH * OW * C;
  if (idx >= total) return;
  int c = idx % C;
  int pp = idx / C;
  int ow = pp % OW, oh = pp / OW;
  float v = (float)a[((long)(oh >> 1) * (OW >> 1) + (ow >> 1)) * C + c];
  if (b) v += (float)b[idx];
  d[idx] = (_Float16)v;
}

__global__ void k_relu_h(const _Float16* __restrict__ x, _Float16* __restrict__ y, int n) {
  int idx = blockIdx.x * blockDim.x + threadIdx.x;
  if (idx >= n) return;
  float v = (float)x[idx];
  y[idx] = (_Float16)fmaxf(v, 0.f);
}

// ---------------- decode (sigmoid + max + bbox transform + clip) ----------------

__global__ void k_decode(const _Float16* __restrict__ regY, const _Float16* __restrict__ clsY,
                         const float* __restrict__ anchors,
                         float* __restrict__ rawScore, float* __restrict__ sWork,
                         int* __restrict__ argCls, float* __restrict__ boxes,
                         float* __restrict__ areas,
                         int P, int regStride, int clsStride, int off) {
  int j = blockIdx.x * blockDim.x + threadIdx.x;
  if (j >= P * 9) return;
  int p = j / 9, a = j - p * 9;
  const _Float16* rp = regY + (long)p * regStride + a * 4;
  float r0 = (float)rp[0], r1 = (float)rp[1], r2 = (float)rp[2], r3 = (float)rp[3];
  const _Float16* cp = clsY + (long)p * clsStride + a * 80;
  float m = -1.f; int am = 0;
  for (int c = 0; c < 80; ++c) {
    float s = 1.f / (1.f + __expf(-(float)cp[c]));
    if (s > m) { m = s; am = c; }
  }
  int gi = off + j;
  const float* an = anchors + (long)gi * 4;
  float wa = an[2] - an[0], ha = an[3] - an[1];
  float cxa = an[0] + 0.5f * wa, cya = an[1] + 0.5f * ha;
  float cx = cxa + r0 * 0.1f * wa;
  float cy = cya + r1 * 0.1f * ha;
  float w = __expf(r2 * 0.2f) * wa;
  float h = __expf(r3 * 0.2f) * ha;
  float x1 = fminf(fmaxf(cx - 0.5f * w, 0.f), 1024.f);
  float y1 = fminf(fmaxf(cy - 0.5f * h, 0.f), 1024.f);
  float x2 = fminf(fmaxf(cx + 0.5f * w, 0.f), 1024.f);
  float y2 = fminf(fmaxf(cy + 0.5f * h, 0.f), 1024.f);
  boxes[(long)gi * 4 + 0] = x1;
  boxes[(long)gi * 4 + 1] = y1;
  boxes[(long)gi * 4 + 2] = x2;
  boxes[(long)gi * 4 + 3] = y2;
  areas[gi] = (x2 - x1) * (y2 - y1);
  rawScore[gi] = m;
  argCls[gi] = am;
  sWork[gi] = (m > 0.05f) ? m : NEGINF;
}

// ---------------- greedy NMS (single workgroup, 100 iterations) ----------------

__global__ __launch_bounds__(512)
void k_nms(float* __restrict__ s, const float* __restrict__ rawScore,
           const int* __restrict__ argCls, const float* __restrict__ boxes,
           const float* __restrict__ areas, int N, float* __restrict__ out) {
  __shared__ float sv[512];
  __shared__ int si[512];
  const int tid = threadIdx.x;
  float* outS = out;        // [100]
  float* outC = out + 100;  // [100]
  float* outB = out + 200;  // [100*4]
  float* outV = out + 600;  // [100]

  for (int it = 0; it < 100; ++it) {
    float best = -2.0e30f;
    int bi = 0x7fffffff;
    for (int j = tid; j < N; j += 512) {
      float v = s[j];
      if (v > best || (v == best && j < bi)) { best = v; bi = j; }
    }
    sv[tid] = best; si[tid] = bi;
    __syncthreads();
    for (int off = 256; off > 0; off >>= 1) {
      if (tid < off) {
        float v2 = sv[tid + off]; int i2 = si[tid + off];
        if (v2 > sv[tid] || (v2 == sv[tid] && i2 < si[tid])) { sv[tid] = v2; si[tid] = i2; }
      }
      __syncthreads();
    }
    int i = si[0];
    float smax = sv[0];
    bool ok = smax > (NEGINF * 0.5f);
    float bx1 = boxes[(long)i * 4 + 0], by1 = boxes[(long)i * 4 + 1];
    float bx2 = boxes[(long)i * 4 + 2], by2 = boxes[(long)i * 4 + 3];
    float ar = areas[i];
    if (tid == 0) {
      float vf = ok ? 1.f : 0.f;
      outS[it] = ok ? rawScore[i] : 0.f;
      outC[it] = ok ? (float)argCls[i] : -1.f;
      outB[it * 4 + 0] = bx1 * vf;
      outB[it * 4 + 1] = by1 * vf;
      outB[it * 4 + 2] = bx2 * vf;
      outB[it * 4 + 3] = by2 * vf;
      outV[it] = vf;
      s[i] = NEGINF;
    }
    for (int j = tid; j < N; j += 512) {
      float xx1 = fmaxf(bx1, boxes[(long)j * 4 + 0]);
      float yy1 = fmaxf(by1, boxes[(long)j * 4 + 1]);
      float xx2 = fminf(bx2, boxes[(long)j * 4 + 2]);
      float yy2 = fminf(by2, boxes[(long)j * 4 + 3]);
      float inter = fmaxf(xx2 - xx1, 0.f) * fmaxf(yy2 - yy1, 0.f);
      float iou = inter / fmaxf(ar + areas[j] - inter, 1e-8f);
      if (iou > 0.5f) s[j] = NEGINF;
    }
    __syncthreads();
  }
}

// ---------------- host orchestration ----------------

extern "C" void kernel_launch(void* const* d_in, const int* in_sizes, int n_in,
                              void* d_out, int out_size, void* d_ws, size_t ws_size,
                              hipStream_t stream) {
  (void)in_sizes; (void)n_in; (void)out_size; (void)ws_size;

  const float* C3 = (const float*)d_in[0];       // [40,64,64]
  const float* C4 = (const float*)d_in[1];       // [112,32,32]
  const float* C5 = (const float*)d_in[2];       // [320,32,32]
  const float* anchors = (const float*)d_in[3];  // [196416,4]

  // params flattened in jax-pytree (alphabetical key) order
  const float* wP3_1 = (const float*)d_in[4];  const float* bP3_1 = (const float*)d_in[5];
  const float* wP3_2 = (const float*)d_in[6];  const float* bP3_2 = (const float*)d_in[7];
  const float* wP4_1 = (const float*)d_in[8];  const float* bP4_1 = (const float*)d_in[9];
  const float* wP4_2 = (const float*)d_in[10]; const float* bP4_2 = (const float*)d_in[11];
  const float* wP5_1 = (const float*)d_in[12]; const float* bP5_1 = (const float*)d_in[13];
  const float* wP5_2 = (const float*)d_in[14]; const float* bP5_2 = (const float*)d_in[15];
  const float* wP6   = (const float*)d_in[16]; const float* bP6   = (const float*)d_in[17];
  const float* wP7_2 = (const float*)d_in[18]; const float* bP7_2 = (const float*)d_in[19];
  const float* wCls[4]; const float* bCls[4];
  for (int i = 0; i < 4; ++i) { wCls[i] = (const float*)d_in[20 + 2 * i]; bCls[i] = (const float*)d_in[21 + 2 * i]; }
  const float* wClsOut = (const float*)d_in[28]; const float* bClsOut = (const float*)d_in[29];
  const float* wReg[4]; const float* bReg[4];
  for (int i = 0; i < 4; ++i) { wReg[i] = (const float*)d_in[30 + 2 * i]; bReg[i] = (const float*)d_in[31 + 2 * i]; }
  const float* wRegOut = (const float*)d_in[38]; const float* bRegOut = (const float*)d_in[39];

  char* cur = (char*)d_ws;
  auto alloc = [&](size_t bytes) -> void* {
    void* p = (void*)cur;
    cur += (bytes + 255) & ~(size_t)255;
    return p;
  };
  auto allocH = [&](size_t n) { return (_Float16*)alloc(n * sizeof(_Float16)); };
  auto allocF = [&](size_t n) { return (float*)alloc(n * sizeof(float)); };

  auto pack = [&](const float* w, int Cout, int Coutp, int Cin, int Cinp, int K) {
    _Float16* o = allocH((size_t)K * K * Coutp * Cinp);
    int total = K * K * Coutp * Cinp;
    k_pack_w<<<dim3((total + 255) / 256), dim3(256), 0, stream>>>(w, o, Cout, Coutp, Cin, Cinp, K);
    return o;
  };
  auto conv = [&](const _Float16* x, const _Float16* wp, const float* bias, _Float16* y,
                  int IH, int IW, int Cinp, int OH, int OW, int Cout, int Coutp,
                  int K, int s, int pd, int relu) {
    dim3 g((OH * OW + 63) / 64, Coutp / 64);
    k_conv_wmma<<<g, dim3(128), 0, stream>>>(x, wp, bias, y, IH, IW, Cinp, OH, OW, Cout, Coutp, K, s, pd, relu);
  };
  auto up2 = [&](const _Float16* a, const _Float16* b, _Float16* d, int OH, int OW) {
    int total = OH * OW * 256;
    k_up2_add<<<(total + 255) / 256, 256, 0, stream>>>(a, b, d, OH, OW, 256);
  };

  // ---- pack all weights (Cin padded to x32, Cout padded to x64) ----
  _Float16* pP3_1 = pack(wP3_1, 256, 256, 40, 64, 1);
  _Float16* pP3_2 = pack(wP3_2, 256, 256, 256, 256, 3);
  _Float16* pP4_1 = pack(wP4_1, 256, 256, 112, 128, 1);
  _Float16* pP4_2 = pack(wP4_2, 256, 256, 256, 256, 3);
  _Float16* pP5_1 = pack(wP5_1, 256, 256, 320, 320, 1);
  _Float16* pP5_2 = pack(wP5_2, 256, 256, 256, 256, 3);
  _Float16* pP6   = pack(wP6,   256, 256, 320, 320, 3);
  _Float16* pP7_2 = pack(wP7_2, 256, 256, 256, 256, 3);
  _Float16* pCls[4]; for (int i = 0; i < 4; ++i) pCls[i] = pack(wCls[i], 256, 256, 256, 256, 3);
  _Float16* pClsOut = pack(wClsOut, 720, 768, 256, 256, 3);
  _Float16* pReg[4]; for (int i = 0; i < 4; ++i) pReg[i] = pack(wReg[i], 256, 256, 256, 256, 3);
  _Float16* pRegOut = pack(wRegOut, 36, 64, 256, 256, 3);

  // ---- convert backbone features to NHWC f16 (channel padded) ----
  _Float16* c3h = allocH((size_t)64 * 64 * 64);
  k_nchw_to_nhwc_h<<<(64 * 64 * 64 + 255) / 256, 256, 0, stream>>>(C3, c3h, 40, 64, 64 * 64);
  _Float16* c4h = allocH((size_t)32 * 32 * 128);
  k_nchw_to_nhwc_h<<<(32 * 32 * 128 + 255) / 256, 256, 0, stream>>>(C4, c4h, 112, 128, 32 * 32);
  _Float16* c5h = allocH((size_t)32 * 32 * 320);
  k_nchw_to_nhwc_h<<<(32 * 32 * 320 + 255) / 256, 256, 0, stream>>>(C5, c5h, 320, 320, 32 * 32);

  // ---- FPN ----
  _Float16* p51o  = allocH((size_t)32 * 32 * 256);
  _Float16* p5u   = allocH((size_t)64 * 64 * 256);
  _Float16* P5    = allocH((size_t)32 * 32 * 256);
  _Float16* p41o  = allocH((size_t)32 * 32 * 256);
  _Float16* p4pre = allocH((size_t)64 * 64 * 256);
  _Float16* p4u   = allocH((size_t)128 * 128 * 256);
  _Float16* P4    = allocH((size_t)64 * 64 * 256);
  _Float16* p31o  = allocH((size_t)64 * 64 * 256);
  _Float16* p3pre = allocH((size_t)128 * 128 * 256);
  _Float16* P3    = allocH((size_t)128 * 128 * 256);
  _Float16* P6    = allocH((size_t)16 * 16 * 256);
  _Float16* P6r   = allocH((size_t)16 * 16 * 256);
  _Float16* P7    = allocH((size_t)8 * 8 * 256);

  conv(c5h, pP5_1, bP5_1, p51o, 32, 32, 320, 32, 32, 256, 256, 1, 1, 0, 0);
  up2(p51o, nullptr, p5u, 64, 64);
  conv(p51o, pP5_2, bP5_2, P5, 32, 32, 256, 32, 32, 256, 256, 3, 1, 1, 0);
  conv(c4h, pP4_1, bP4_1, p41o, 32, 32, 128, 32, 32, 256, 256, 1, 1, 0, 0);
  up2(p41o, p5u, p4pre, 64, 64);
  up2(p4pre, nullptr, p4u, 128, 128);
  conv(p4pre, pP4_2, bP4_2, P4, 64, 64, 256, 64, 64, 256, 256, 3, 1, 1, 0);
  conv(c3h, pP3_1, bP3_1, p31o, 64, 64, 64, 64, 64, 256, 256, 1, 1, 0, 0);
  up2(p31o, p4u, p3pre, 128, 128);
  conv(p3pre, pP3_2, bP3_2, P3, 128, 128, 256, 128, 128, 256, 256, 3, 1, 1, 0);
  conv(c5h, pP6, bP6, P6, 32, 32, 320, 16, 16, 256, 256, 3, 2, 1, 0);
  k_relu_h<<<(16 * 16 * 256 + 255) / 256, 256, 0, stream>>>(P6, P6r, 16 * 16 * 256);
  conv(P6r, pP7_2, bP7_2, P7, 16, 16, 256, 8, 8, 256, 256, 3, 2, 1, 0);

  // ---- heads + decode, level by level ----
  _Float16* hA = allocH((size_t)16384 * 256);
  _Float16* hB = allocH((size_t)16384 * 256);
  _Float16* regBuf = allocH((size_t)16384 * 64);
  _Float16* clsBuf = allocH((size_t)16384 * 768);

  const int N = 196416;  // total anchors
  float* rawScore = allocF(N);
  float* sWork = allocF(N);
  float* areas = allocF(N);
  float* boxesAll = allocF((size_t)N * 4);
  int* argCls = (int*)alloc((size_t)N * sizeof(int));

  const _Float16* feats[5] = {P3, P4, P5, P6, P7};
  const int sizes[5] = {128, 64, 32, 16, 8};
  int off = 0;
  for (int L = 0; L < 5; ++L) {
    int S = sizes[L];
    int P = S * S;
    const _Float16* f = feats[L];
    // regression head
    conv(f,  pReg[0], bReg[0], hA, S, S, 256, S, S, 256, 256, 3, 1, 1, 1);
    conv(hA, pReg[1], bReg[1], hB, S, S, 256, S, S, 256, 256, 3, 1, 1, 1);
    conv(hB, pReg[2], bReg[2], hA, S, S, 256, S, S, 256, 256, 3, 1, 1, 1);
    conv(hA, pReg[3], bReg[3], hB, S, S, 256, S, S, 256, 256, 3, 1, 1, 1);
    conv(hB, pRegOut, bRegOut, regBuf, S, S, 256, S, S, 36, 64, 3, 1, 1, 0);
    // classification head
    conv(f,  pCls[0], bCls[0], hA, S, S, 256, S, S, 256, 256, 3, 1, 1, 1);
    conv(hA, pCls[1], bCls[1], hB, S, S, 256, S, S, 256, 256, 3, 1, 1, 1);
    conv(hB, pCls[2], bCls[2], hA, S, S, 256, S, S, 256, 256, 3, 1, 1, 1);
    conv(hA, pCls[3], bCls[3], hB, S, S, 256, S, S, 256, 256, 3, 1, 1, 1);
    conv(hB, pClsOut, bClsOut, clsBuf, S, S, 256, S, S, 720, 768, 3, 1, 1, 0);
    // decode this level
    int na = P * 9;
    k_decode<<<(na + 255) / 256, 256, 0, stream>>>(regBuf, clsBuf, anchors, rawScore, sWork,
                                                   argCls, boxesAll, areas, P, 64, 768, off);
    off += na;
  }

  // ---- greedy NMS -> final outputs [scores 100 | class 100 | boxes 400 | valid 100] ----
  k_nms<<<1, 512, 0, stream>>>(sWork, rawScore, argCls, boxesAll, areas, N, (float*)d_out);
}
